// GraphAttentionLayer_75436805587418
// MI455X (gfx1250) — matrix operand order repaired and checked by reading, not compiled, not run on hardware
//
#include <hip/hip_runtime.h>
#include <hip/hip_bf16.h>

// ---------------------------------------------------------------------------
// GAT layer, fused flash-attention style, CDNA5 (gfx1250) wave32 + WMMA f16.
//   Phase 0: ballot-pack adj [N,N] i32 -> bitmask [N, N/32] u32 (2 MB).
//   Phase 1: project h -> Whs (f16), Wht (f16), Whc^T (f16) in workspace.
//   Phase 2: per-wave 16-row tile, 64-key steps:
//     - 8x v_wmma_f32_16x16x32_f16 for scores S (K=64, 4 col subtiles)
//     - online softmax with a skip-path: max butterfly + rescale only run on
//       the rare iterations where some row's running max changes
//     - masking applied as p=0 after exp (identical to -9e15 in f32)
//     - row sums via "P @ ones" WMMA tile (no shuffle reduction)
//     - P -> A-layout via per-wave LDS bounce
//     - 18x WMMA: out += P @ Whc (8 col tiles x 2 K-chunks) + 2 ones-WMMAs
// ---------------------------------------------------------------------------

typedef _Float16 half_t;
typedef __attribute__((ext_vector_type(8)))  _Float16 v8h;
typedef __attribute__((ext_vector_type(16))) _Float16 v16h;
typedef __attribute__((ext_vector_type(8)))  float    v8f;

#define GAT_B    4
#define GAT_N    4096
#define GAT_DIN  128
#define GAT_DH   64
#define GAT_DOUT 128
#define GAT_NW   (GAT_N / 32)       // 128 u32 words per adjacency row

// 16 contiguous halfs -> A/B fragment in this file's contiguous k = 16h + s order.
__device__ __forceinline__ v16h ld16_contig(const half_t* p) {
  v8h lo = *(const v8h*)(p);
  v8h hi = *(const v8h*)(p + 8);
  return __builtin_shufflevector(lo, hi, 0,1,2,3,4,5,6,7,8,9,10,11,12,13,14,15);
}

// A-matrix lane pattern: halfs [0..7] and [16..23] relative to p.
__device__ __forceinline__ v16h ld16_afrag(const half_t* p) {
  v8h lo = *(const v8h*)(p);
  v8h hi = *(const v8h*)(p + 16);
  return __builtin_shufflevector(lo, hi, 0,1,2,3,4,5,6,7,8,9,10,11,12,13,14,15);
}

__device__ __forceinline__ v8f wmma_f16(v16h a, v16h b, v8f c) {
  // (neg_a, A, neg_b, B, c_mod, C, reuse_a, reuse_b)
  return __builtin_amdgcn_wmma_f32_16x16x32_f16(false, a, false, b, (short)0, c,
                                                false, false);
}

// ---------------------------------------------------------------------------
// Phase 0: pack adjacency to bits. One wave covers one 32-column group.
// ---------------------------------------------------------------------------
__global__ __launch_bounds__(256)
void gat_pack_adj(const int* __restrict__ adj, unsigned* __restrict__ adjb) {
  const size_t idx = (size_t)blockIdx.x * 256 + threadIdx.x;  // element id
  const unsigned m = __builtin_amdgcn_ballot_w32(adj[idx] > 0);
  if ((threadIdx.x & 31) == 0) adjb[idx >> 5] = m;
}

// ---------------------------------------------------------------------------
// Phase 1: projections. One block per (b, n) row; 128 threads.
//   Whs  [B*N, 64]  f16   (row-major)
//   Wht  [B*N, 64]  f16   (row-major)
//   Whct [B, 128, N] f16  (transposed: column c contiguous over n)
// ---------------------------------------------------------------------------
__global__ __launch_bounds__(128)
void gat_proj_kernel(const float* __restrict__ h,
                     const float* __restrict__ Ws,
                     const float* __restrict__ Wt,
                     const float* __restrict__ Wc,
                     half_t* __restrict__ Whs,
                     half_t* __restrict__ Wht,
                     half_t* __restrict__ Whct) {
  const int row = blockIdx.x;            // 0 .. B*N-1
  const int b   = row >> 12;
  const int n   = row & (GAT_N - 1);
  const int t   = threadIdx.x;           // 0..127

  __shared__ float hrow[GAT_DIN];
  hrow[t] = h[(size_t)row * GAT_DIN + t];
  __syncthreads();

  float accC = 0.0f;
#pragma unroll 8
  for (int i = 0; i < GAT_DIN; ++i) accC += hrow[i] * Wc[i * GAT_DOUT + t];
  Whct[((size_t)b * GAT_DOUT + t) * GAT_N + n] = (half_t)accC;

  if (t < GAT_DH) {
    float accS = 0.0f, accT = 0.0f;
#pragma unroll 8
    for (int i = 0; i < GAT_DIN; ++i) {
      const float hv = hrow[i];
      accS += hv * Ws[i * GAT_DH + t];
      accT += hv * Wt[i * GAT_DH + t];
    }
    Whs[(size_t)row * GAT_DH + t] = (half_t)accS;
    Wht[(size_t)row * GAT_DH + t] = (half_t)accT;
  }
}

// ---------------------------------------------------------------------------
// Phase 2: fused masked-softmax attention. One wave per block, 16 query rows.
// blockIdx.x = row tile (N/16), blockIdx.y = batch.
// ---------------------------------------------------------------------------
__global__ __launch_bounds__(32)
void gat_attn_kernel(const half_t*   __restrict__ Whs,
                     const half_t*   __restrict__ Wht,
                     const half_t*   __restrict__ Whct,
                     const unsigned* __restrict__ adjb,
                     float*          __restrict__ out) {
  const int b      = blockIdx.y;
  const int r0     = blockIdx.x * 16;
  const int lane   = threadIdx.x;
  const int lmod   = lane & 15;
  const int hiHalf = (lane >> 4) & 1;     // lanes 16..31
  const int rowOfs = hiHalf ? 8 : 0;      // C-layout: lane group -> row offset
  const int kb     = hiHalf ? 16 : 0;     // A-layout lane k-base

  __shared__ __align__(16) half_t Pbuf[16][64];   // P tile (16 rows x 64 keys)

  // per-lane adjacency bit selectors: subtile parity picks bit lmod or lmod+16
  const unsigned bmLo = 1u << lmod;
  const unsigned bmHi = 1u << (lmod + 16);

  // ---- A fragments for S = Whs @ Wht^T : rows r0..r0+15, K = 64
  const half_t* whsRow = Whs + ((size_t)(b * GAT_N + r0 + lmod)) * GAT_DH;
  const v16h a0 = ld16_contig(whsRow + 0  + kb);
  const v16h a1 = ld16_contig(whsRow + 32 + kb);

  const half_t* whtBase  = Wht  + (size_t)b * GAT_N * GAT_DH;
  const half_t* whctBase = Whct + ((size_t)b * GAT_DOUT + lmod) * GAT_N;

  // constant all-ones B fragment (for row-sum WMMA)
  v16h ones;
#pragma unroll
  for (int i = 0; i < 16; ++i) ones[i] = (_Float16)1.0f;

  // ---- flash-softmax state: acc[0..7]=output tiles, acc[8]=row-sum tile
  float m_run[8];
  v8f acc[9];
  const v8f vzero = {};
#pragma unroll
  for (int v = 0; v < 8; ++v) m_run[v] = -3.0e38f;
#pragma unroll
  for (int c = 0; c < 9; ++c) acc[c] = vzero;

  for (int j = 0; j < GAT_N; j += 64) {
    // ---- scores for 4 subtiles of 16 keys (K = 64 contraction)
    v8f s[4];
#pragma unroll
    for (int sub = 0; sub < 4; ++sub) {
      const int js = j + sub * 16;
      const half_t* wr =
          whtBase + (size_t)(js + lmod) * GAT_DH + (hiHalf ? 16 : 0);
      v8f t = vzero;
      t = wmma_f16(a0, ld16_contig(wr),      t);   // K = 0..31
      t = wmma_f16(a1, ld16_contig(wr + 32), t);   // K = 32..63
      s[sub] = t;
    }

    // ---- adjacency words (2 per row per 64 keys)
    const unsigned* adjRow =
        adjb + (size_t)(r0 + rowOfs) * GAT_NW + (j >> 5);
    unsigned w0[8], w1[8];
#pragma unroll
    for (int v = 0; v < 8; ++v) {
      w0[v] = adjRow[(size_t)v * GAT_NW];
      w1[v] = adjRow[(size_t)v * GAT_NW + 1];
    }

    // ---- lane-local max of raw scores; detect whether any row max grows.
    // (Using the unmasked max is valid: the common exp(-m) factor cancels
    //  in p/sum(p); masked elements become exactly 0 below.)
    float lmax[8];
    bool updL = false;
#pragma unroll
    for (int v = 0; v < 8; ++v) {
      lmax[v] = fmaxf(fmaxf(s[0][v], s[1][v]), fmaxf(s[2][v], s[3][v]));
      updL = updL || (lmax[v] > m_run[v]);
    }

    if (__any(updL)) {   // wave-uniform slow path: new max somewhere
      float rowScale[8];
#pragma unroll
      for (int v = 0; v < 8; ++v) {
        float mx = lmax[v];
        mx = fmaxf(mx, __shfl_xor(mx, 1, 32));
        mx = fmaxf(mx, __shfl_xor(mx, 2, 32));
        mx = fmaxf(mx, __shfl_xor(mx, 4, 32));
        mx = fmaxf(mx, __shfl_xor(mx, 8, 32));
        const float m_new = fmaxf(m_run[v], mx);
        rowScale[v] = __expf(m_run[v] - m_new);
        m_run[v]    = m_new;
      }
#pragma unroll
      for (int c = 0; c < 9; ++c)
#pragma unroll
        for (int v = 0; v < 8; ++v) acc[c][v] *= rowScale[v];
    }

    // ---- probabilities: p = adj ? exp(s - m) : 0 ; f16 -> LDS scatter
#pragma unroll
    for (int v = 0; v < 8; ++v) {
      float p0 = __expf(s[0][v] - m_run[v]);
      float p1 = __expf(s[1][v] - m_run[v]);
      float p2 = __expf(s[2][v] - m_run[v]);
      float p3 = __expf(s[3][v] - m_run[v]);
      p0 = (w0[v] & bmLo) ? p0 : 0.0f;
      p1 = (w0[v] & bmHi) ? p1 : 0.0f;
      p2 = (w1[v] & bmLo) ? p2 : 0.0f;
      p3 = (w1[v] & bmHi) ? p3 : 0.0f;
      Pbuf[v + rowOfs][lmod]      = (half_t)p0;
      Pbuf[v + rowOfs][16 + lmod] = (half_t)p1;
      Pbuf[v + rowOfs][32 + lmod] = (half_t)p2;
      Pbuf[v + rowOfs][48 + lmod] = (half_t)p3;
    }

    // same-wave cross-lane LDS visibility before b128 reads
    asm volatile("s_wait_dscnt 0" ::: "memory");

    // ---- P as two A fragments (16 rows x K=32 each)
    const v16h pa0 = ld16_contig(&Pbuf[lmod][kb]);
    const v16h pa1 = ld16_contig(&Pbuf[lmod][32 + kb]);

    // ---- out += P @ Whc[j..j+63, :], 8 column tiles x 2 K-chunks
    const half_t* wcB = whctBase + j + (hiHalf ? 16 : 0);
#pragma unroll
    for (int c = 0; c < 8; ++c) {
      const half_t* pc = wcB + (size_t)c * 16 * GAT_N;
      acc[c] = wmma_f16(pa0, ld16_contig(pc),      acc[c]);
      acc[c] = wmma_f16(pa1, ld16_contig(pc + 32), acc[c]);
    }
    // ---- row sums via matrix unit: acc8 += P @ ones
    acc[8] = wmma_f16(pa0, ones, acc[8]);
    acc[8] = wmma_f16(pa1, ones, acc[8]);
  }

  // ---- epilogue: divide by row sums (acc[8]), store f32 output
  float inv[8];
#pragma unroll
  for (int v = 0; v < 8; ++v) inv[v] = 1.0f / acc[8][v];

  float* outBase = out + (size_t)(b * GAT_N + r0 + rowOfs) * GAT_DOUT;
#pragma unroll
  for (int c = 0; c < 8; ++c) {
#pragma unroll
    for (int v = 0; v < 8; ++v) {
      outBase[(size_t)v * GAT_DOUT + c * 16 + lmod] = acc[c][v] * inv[v];
    }
  }
}

// ---------------------------------------------------------------------------
extern "C" void kernel_launch(void* const* d_in, const int* in_sizes, int n_in,
                              void* d_out, int out_size, void* d_ws,
                              size_t ws_size, hipStream_t stream) {
  const float* h   = (const float*)d_in[0];   // [B,N,128] f32
  const int*   adj = (const int*)  d_in[1];   // [N,N] i32
  const float* Ws  = (const float*)d_in[2];   // [128,64] f32
  const float* Wt  = (const float*)d_in[3];   // [128,64] f32
  const float* Wc  = (const float*)d_in[4];   // [128,128] f32
  float*       out = (float*)d_out;           // [B,N,128] f32

  half_t*   whs  = (half_t*)d_ws;                                // 2 MB
  half_t*   wht  = whs + (size_t)GAT_B * GAT_N * GAT_DH;         // 2 MB
  half_t*   whct = wht + (size_t)GAT_B * GAT_N * GAT_DH;         // 4 MB
  unsigned* adjb = (unsigned*)(whct + (size_t)GAT_B * GAT_DOUT * GAT_N); // 2 MB

  gat_pack_adj<<<dim3((unsigned)(((size_t)GAT_N * GAT_N) / 256)), 256, 0,
                 stream>>>(adj, adjb);

  gat_proj_kernel<<<dim3(GAT_B * GAT_N), 128, 0, stream>>>(
      h, Ws, Wt, Wc, whs, wht, whct);

  gat_attn_kernel<<<dim3(GAT_N / 16, GAT_B), 32, 0, stream>>>(
      whs, wht, whct, adjb, out);
}
